// SceneSayerODE_86973087744262
// MI455X (gfx1250) — matrix operand order repaired and checked by reading, not compiled, not run on hardware
//
#include <hip/hip_runtime.h>
#include <math.h>

// ---------------------------------------------------------------------------
// SceneSayer transformer forward for MI455X (gfx1250, wave32, WMMA).
// All large matmuls use v_wmma_f32_16x16x32_f16 with f32 accumulation.
// Weights are converted once per call to f16 transposed (N x Kpad) layout so
// both A and B WMMA fragments are contiguous 16B global loads that match the
// documented CDNA5 16-bit A(16x32)/B(32x16) VGPR layouts.
//
// d_in layout (jax pytree flatten: top-level dict insertion order, params
// dict key-sorted, lists in order):
//  0 features 1 pair_idx 2 union_feat 3 spatial_masks 4 pred_labels 5 boxes
//  6 a_rel.b 7 a_rel.w 8-11 bn1{b,g,m,v} 12-15 bn2{b,g,m,v}
//  16 box1.b 17 box1.w 18 box2.b 19 box2.w 20 c_rel.b 21 c_rel.w
//  22 conv1_b 23 conv1_w 24 conv2_b 25 conv2_w
//  26..37 global layer0, 38..49 layer1, 50..61 layer2, 62..73 local layer
//   (per layer: ff1.b ff1.w ff2.b ff2.w ln1_b ln1_g ln2_b ln2_g out.b out.w qkv.b qkv.w)
//  74 obj_embed 75 obj_embed2 76 obj_fc.b 77 obj_fc.w 78 s_rel.b 79 s_rel.w
//  80 subj_fc.b 81 subj_fc.w 82 union_b 83 union_w 84 vr_fc.b 85 vr_fc.w
// ---------------------------------------------------------------------------

typedef _Float16 h16;
typedef __attribute__((ext_vector_type(8)))  _Float16 v8h;
typedef __attribute__((ext_vector_type(16))) _Float16 v16h;
typedef __attribute__((ext_vector_type(8)))  float    v8f;

#define DREAL  1936
#define DP     1984   // D padded to 64
#define NHEAD  8
#define HDIM   242
#define FFN    2048
#define NPAIR  1024
#define NFRAME 128
#define PPF    8
#define QKVN   5808
#define QKVP   5824   // padded to 64
#define KFEAT  2376
#define KFEATP 2400   // padded to 32
#define UROWS  50176  // 1024*49 (multiple of 128)

__device__ __forceinline__ float dsigmoid(float x) { return 1.0f / (1.0f + __expf(-x)); }

// Load a 16-element f16 fragment as two contiguous 16B chunks separated by
// `gap` halves (A fragment: gap=16, B fragment: gap=16).
__device__ __forceinline__ v16h ld_frag(const h16* __restrict__ p, int gap)
{
  v8h lo = *(const v8h*)p;
  v8h hi = *(const v8h*)(p + gap);
  return __builtin_shufflevector(lo, hi, 0,1,2,3,4,5,6,7,8,9,10,11,12,13,14,15);
}

__device__ __forceinline__ v8f wmma_f16(v16h a, v16h b, v8f c)
{
  return __builtin_amdgcn_wmma_f32_16x16x32_f16(false, a, false, b, (short)0, c,
                                                false, false);
}

// ---------------- WMMA GEMM: C[M,Np] = A[M,Kp] * Bt[Np,Kp]^T (+bias, act) ---
// A row-major f16 (ld = Kp), Bt = weight stored transposed f16 (ld = Kp).
// Block: 128 threads = 4 waves stacked in M; block tile 128x64.
// Wave tile 32x64 = 8 WMMAs per k-step on 12 b128 loads, 2-stage software
// pipeline: all loads for k+32 are issued before the 8 WMMAs for k, so the
// inner loop is one load clause followed by stall-free WMMAs.
// act: 0=none, 1=relu.  Requires M % 128 == 0, Np % 64 == 0, Kp % 32 == 0.
__global__ __launch_bounds__(128)
void k_gemm(const h16* __restrict__ A, const h16* __restrict__ Bt,
            float* __restrict__ C32, h16* __restrict__ C16,
            const float* __restrict__ bias,
            int Kp, int Np, int Nreal, int act)
{
  const int lane = threadIdx.x & 31;
  const int wave = threadIdx.x >> 5;
  const int m0 = blockIdx.y * 128 + wave * 32;
  const int n0 = blockIdx.x * 64;

  const int loffA = (lane >> 4) * 8;    // A K sub-offset per half-wave
  const int koffB = (lane >> 4) * 8;    // B K sub-offset per half-wave

  const h16* Ap0 = A + (size_t)(m0 + (lane & 15)) * Kp + loffA;
  const h16* Ap1 = Ap0 + (size_t)16 * Kp;
  const h16* Bp  = Bt + (size_t)(n0 + (lane & 15)) * Kp + koffB;

  v8f acc[2][4] = {};

  // stage registers (current k-step fragments)
  v16h a0 = ld_frag(Ap0, 16);
  v16h a1 = ld_frag(Ap1, 16);
  v16h b[4];
#pragma unroll
  for (int t = 0; t < 4; ++t) b[t] = ld_frag(Bp + (size_t)(t * 16) * Kp, 16);

  for (int k0 = 0; k0 + 32 < Kp; k0 += 32) {
    const int k1 = k0 + 32;
    // issue next-stage loads first (no consumer until rotation)
    v16h na0 = ld_frag(Ap0 + k1, 16);
    v16h na1 = ld_frag(Ap1 + k1, 16);
    v16h nb[4];
#pragma unroll
    for (int t = 0; t < 4; ++t) nb[t] = ld_frag(Bp + (size_t)(t * 16) * Kp + k1, 16);
    // compute on current stage (data arrived during previous iteration)
#pragma unroll
    for (int t = 0; t < 4; ++t) {
      acc[0][t] = wmma_f16(a0, b[t], acc[0][t]);
      acc[1][t] = wmma_f16(a1, b[t], acc[1][t]);
    }
    // rotate
    a0 = na0; a1 = na1;
#pragma unroll
    for (int t = 0; t < 4; ++t) b[t] = nb[t];
  }
  // epilogue: last k-step
#pragma unroll
  for (int t = 0; t < 4; ++t) {
    acc[0][t] = wmma_f16(a0, b[t], acc[0][t]);
    acc[1][t] = wmma_f16(a1, b[t], acc[1][t]);
  }

#pragma unroll
  for (int i = 0; i < 2; ++i) {
    const int rbase = m0 + i * 16 + ((lane >> 4) * 8);
#pragma unroll
    for (int t = 0; t < 4; ++t) {
      int col = n0 + t * 16 + (lane & 15);
      float bv = (bias && col < Nreal) ? bias[col] : 0.0f;
#pragma unroll
      for (int r = 0; r < 8; ++r) {
        float v = acc[i][t][r] + bv;
        if (act == 1) v = fmaxf(v, 0.0f);
        size_t o = (size_t)(rbase + r) * Np + col;
        if (C32) C32[o] = v;
        if (C16) C16[o] = (h16)v;
      }
    }
  }
}

// ---------------- converters ----------------------------------------------
// W (K,N) f32 -> Bt (Np,Kp) f16, zero padded.
__global__ void k_cvt_wT(const float* __restrict__ W, h16* __restrict__ Bt,
                         int K, int N, int Kp, int Np)
{
  size_t idx = (size_t)blockIdx.x * blockDim.x + threadIdx.x;
  if (idx >= (size_t)Np * Kp) return;
  int k = (int)(idx % Kp), n = (int)(idx / Kp);
  float v = (n < N && k < K) ? W[(size_t)k * N + n] : 0.0f;
  Bt[idx] = (h16)v;
}

// straight f32 -> f16 copy (already (N,K) layout, K multiple of 32)
__global__ void k_cvt_copy(const float* __restrict__ W, h16* __restrict__ Bt, size_t n)
{
  size_t idx = (size_t)blockIdx.x * blockDim.x + threadIdx.x;
  if (idx < n) Bt[idx] = (h16)W[idx];
}

// gather feature rows via pair_idx[:,sel] -> f16 (NPAIR, KFEATP)
__global__ void k_gather(const float* __restrict__ F, const int* __restrict__ pidx,
                         int sel, h16* __restrict__ A)
{
  size_t idx = (size_t)blockIdx.x * blockDim.x + threadIdx.x;
  if (idx >= (size_t)NPAIR * KFEATP) return;
  int k = (int)(idx % KFEATP), r = (int)(idx / KFEATP);
  int src = pidx[r * 2 + sel];
  A[idx] = (h16)((k < KFEAT) ? F[(size_t)src * KFEAT + k] : 0.0f);
}

// union_feat (n,c,7,7) -> A16[(n*49+hw), c]
__global__ void k_perm_union(const float* __restrict__ U, h16* __restrict__ A)
{
  size_t idx = (size_t)blockIdx.x * blockDim.x + threadIdx.x;
  if (idx >= (size_t)UROWS * 1024) return;
  int c = (int)(idx % 1024); int r = (int)(idx / 1024);
  int n = r / 49, hw = r % 49;
  A[idx] = (h16)U[((size_t)n * 1024 + c) * 49 + hw];
}

// ---------------- conv path ------------------------------------------------
__global__ void k_conv1(const float* __restrict__ X, const float* __restrict__ W,
                        const float* __restrict__ bias,
                        const float* __restrict__ g, const float* __restrict__ b,
                        const float* __restrict__ m, const float* __restrict__ v,
                        float* __restrict__ O)
{
  int idx = blockIdx.x * blockDim.x + threadIdx.x;
  if (idx >= 1024 * 128 * 196) return;
  int ox = idx % 14, oy = (idx / 14) % 14, oc = (idx / 196) % 128, n = idx / (196 * 128);
  float acc = bias[oc];
  for (int ic = 0; ic < 2; ++ic)
    for (int ky = 0; ky < 7; ++ky) {
      int iy = oy * 2 - 3 + ky; if ((unsigned)iy >= 27u) continue;
      for (int kx = 0; kx < 7; ++kx) {
        int ix = ox * 2 - 3 + kx; if ((unsigned)ix >= 27u) continue;
        acc += X[(((size_t)n * 2 + ic) * 27 + iy) * 27 + ix] *
               W[(((size_t)oc * 2 + ic) * 7 + ky) * 7 + kx];
      }
    }
  acc = fmaxf(acc, 0.0f);
  acc = (acc - m[oc]) * rsqrtf(v[oc] + 1e-5f) * g[oc] + b[oc];
  O[idx] = acc;
}

__global__ void k_pool(const float* __restrict__ I, float* __restrict__ O)
{
  int idx = blockIdx.x * blockDim.x + threadIdx.x;
  if (idx >= 1024 * 128 * 49) return;
  int px = idx % 7, py = (idx / 7) % 7, oc = (idx / 49) % 128, n = idx / (49 * 128);
  float mx = -3.4e38f;
  for (int ky = 0; ky < 3; ++ky) {
    int iy = py * 2 - 1 + ky; if ((unsigned)iy >= 14u) continue;
    for (int kx = 0; kx < 3; ++kx) {
      int ix = px * 2 - 1 + kx; if ((unsigned)ix >= 14u) continue;
      mx = fmaxf(mx, I[(((size_t)n * 128 + oc) * 14 + iy) * 14 + ix]);
    }
  }
  O[idx] = mx;
}

__global__ void k_im2col(const float* __restrict__ P, h16* __restrict__ C)
{
  size_t idx = (size_t)blockIdx.x * blockDim.x + threadIdx.x;
  if (idx >= (size_t)UROWS * 1152) return;
  int kk = (int)(idx % 1152); int r = (int)(idx / 1152);
  int n = r / 49, hw = r % 49, oy = hw / 7, ox = hw % 7;
  int ic = kk / 9, t = kk % 9, ky = t / 3, kx = t % 3;
  int iy = oy - 1 + ky, ix = ox - 1 + kx;
  float vv = ((unsigned)iy < 7u && (unsigned)ix < 7u)
                 ? P[(((size_t)n * 128 + ic) * 7 + iy) * 7 + ix] : 0.0f;
  C[idx] = (h16)vv;
}

// vrin[n, o*49+hw] = bn2(relu(conv2)) + u   (f16 GEMM input)
__global__ void k_vrin(const float* __restrict__ U, const float* __restrict__ C2,
                       const float* __restrict__ g, const float* __restrict__ b,
                       const float* __restrict__ m, const float* __restrict__ v,
                       h16* __restrict__ O)
{
  size_t idx = (size_t)blockIdx.x * blockDim.x + threadIdx.x;
  if (idx >= (size_t)1024 * 256 * 49) return;
  int hw = (int)(idx % 49); int o = (int)((idx / 49) % 256); int n = (int)(idx / (49 * 256));
  size_t r = (size_t)n * 49 + hw;
  float x = C2[r * 256 + o];
  x = fmaxf(x, 0.0f);
  x = (x - m[o]) * rsqrtf(v[o] + 1e-5f) * g[o] + b[o];
  O[(size_t)n * 12544 + (size_t)o * 49 + hw] = (h16)(x + U[r * 256 + o]);
}

// ---------------- concat: rel = [subj|obj|vr|sem] --------------------------
__global__ void k_concat(const float* __restrict__ S, const float* __restrict__ Ob,
                         const float* __restrict__ V,
                         const float* __restrict__ E1, const float* __restrict__ E2,
                         const int* __restrict__ lbl, const int* __restrict__ pidx,
                         float* __restrict__ O32, h16* __restrict__ O16)
{
  size_t idx = (size_t)blockIdx.x * blockDim.x + threadIdx.x;
  if (idx >= (size_t)NPAIR * DP) return;
  int c = (int)(idx % DP); int n = (int)(idx / DP);
  float v = 0.0f;
  if      (c < 512)  v = S [(size_t)n * 512 + c];
  else if (c < 1024) v = Ob[(size_t)n * 512 + (c - 512)];
  else if (c < 1536) v = V [(size_t)n * 512 + (c - 1024)];
  else if (c < 1736) v = E1[(size_t)lbl[pidx[n * 2 + 0]] * 200 + (c - 1536)];
  else if (c < 1936) v = E2[(size_t)lbl[pidx[n * 2 + 1]] * 200 + (c - 1736)];
  O32[idx] = v; O16[idx] = (h16)v;
}

// ---------------- attention (scores+softmax, then A@V) ---------------------
__global__ void k_attn_sm(const float* __restrict__ qkv, float* __restrict__ aw,
                          int S, int causal)
{
  __shared__ float sm[128];
  int q = blockIdx.x % S, bh = blockIdx.x / S;
  int h = bh % NHEAD, b = bh / NHEAD;
  int k = threadIdx.x;
  const float* qv = qkv + (size_t)(b * S + q) * QKVP + h * HDIM;
  const float* kv = qkv + (size_t)(b * S + k) * QKVP + DREAL + h * HDIM;
  float s = 0.0f;
  for (int d = 0; d < HDIM; ++d) s += qv[d] * kv[d];
  s *= 0.064282435f;  // 1/sqrt(242)
  if (causal && k > q) s -= 1e9f;
  sm[k] = s; __syncthreads();
  float mx = -3.4e38f;
  for (int i = 0; i < S; ++i) mx = fmaxf(mx, sm[i]);
  __syncthreads();
  float e = __expf(s - mx);
  sm[k] = e; __syncthreads();
  float sum = 0.0f;
  for (int i = 0; i < S; ++i) sum += sm[i];
  aw[((size_t)bh * S + q) * S + k] = e / sum;
}

__global__ void k_attn_av(const float* __restrict__ qkv, const float* __restrict__ aw,
                          int S, float* __restrict__ O32, h16* __restrict__ O16)
{
  size_t idx = (size_t)blockIdx.x * blockDim.x + threadIdx.x;
  if (idx >= (size_t)NPAIR * DP) return;
  int c = (int)(idx % DP); int row = (int)(idx / DP);
  float o = 0.0f;
  if (c < DREAL) {
    int b = row / S, q = row % S, h = c / HDIM;
    const float* a  = aw + ((size_t)(b * NHEAD + h) * S + q) * S;
    const float* vb = qkv + (size_t)(b * S) * QKVP + 2 * DREAL + c;
    for (int k = 0; k < S; ++k) o += a[k] * vb[(size_t)k * QKVP];
  }
  O32[idx] = o; O16[idx] = (h16)o;
}

// ---------------- residual add + layernorm ---------------------------------
__global__ __launch_bounds__(256)
void k_add_ln(const float* __restrict__ x, const float* __restrict__ y,
              const float* __restrict__ g, const float* __restrict__ b,
              float* __restrict__ o32, h16* __restrict__ o16)
{
  __shared__ float vbuf[DREAL];
  __shared__ float red[256];
  int row = blockIdx.x, tid = threadIdx.x;
  const float* xr = x + (size_t)row * DP;
  const float* yr = y + (size_t)row * DP;
  float s = 0.0f;
  for (int c = tid; c < DREAL; c += 256) { float v = xr[c] + yr[c]; vbuf[c] = v; s += v; }
  red[tid] = s; __syncthreads();
  for (int st = 128; st > 0; st >>= 1) { if (tid < st) red[tid] += red[tid + st]; __syncthreads(); }
  float mean = red[0] / (float)DREAL; __syncthreads();
  s = 0.0f;
  for (int c = tid; c < DREAL; c += 256) { float d = vbuf[c] - mean; s += d * d; }
  red[tid] = s; __syncthreads();
  for (int st = 128; st > 0; st >>= 1) { if (tid < st) red[tid] += red[tid + st]; __syncthreads(); }
  float rstd = rsqrtf(red[0] / (float)DREAL + 1e-5f);
  for (int c = tid; c < DP; c += 256) {
    float v = (c < DREAL) ? (vbuf[c] - mean) * rstd * g[c] + b[c] : 0.0f;
    o32[(size_t)row * DP + c] = v; o16[(size_t)row * DP + c] = (h16)v;
  }
}

// ---------------- transposes (+ positional encoding) -----------------------
__global__ void k_tr_pe(const float* __restrict__ in, float* __restrict__ o32, h16* __restrict__ o16)
{
  size_t idx = (size_t)blockIdx.x * blockDim.x + threadIdx.x;
  if (idx >= (size_t)NPAIR * DP) return;
  int c = (int)(idx % DP); int orow = (int)(idx / DP);
  int p = orow / NFRAME, f = orow % NFRAME;
  float v = 0.0f;
  if (c < DREAL) {
    int i2 = c & ~1;
    float div = __expf((float)i2 * (-9.210340372f / 1936.0f));
    float ang = (float)f * div;
    float pe = (c & 1) ? __cosf(ang) : __sinf(ang);
    v = in[(size_t)(f * PPF + p) * DP + c] + pe;
  }
  o32[idx] = v; o16[idx] = (h16)v;
}

__global__ void k_tr_back(const float* __restrict__ in, float* __restrict__ o32, h16* __restrict__ o16)
{
  size_t idx = (size_t)blockIdx.x * blockDim.x + threadIdx.x;
  if (idx >= (size_t)NPAIR * DP) return;
  int c = (int)(idx % DP); int orow = (int)(idx / DP);
  int f = orow / PPF, p = orow % PPF;
  float v = in[(size_t)(p * NFRAME + f) * DP + c];
  o32[idx] = v; o16[idx] = (h16)v;
}

// ---------------- heads -----------------------------------------------------
__global__ void k_heads(const float* __restrict__ g, const float* __restrict__ bx,
                        const float* __restrict__ aw, const float* __restrict__ ab,
                        const float* __restrict__ sw, const float* __restrict__ sb,
                        const float* __restrict__ cw, const float* __restrict__ cb,
                        const float* __restrict__ b2w, const float* __restrict__ b2b,
                        float* __restrict__ out)
{
  int idx = blockIdx.x * blockDim.x + threadIdx.x;
  if (idx >= NPAIR * 30) return;
  int j = idx % 30, n = idx / 30;
  const float* gr = g + (size_t)n * DP;
  if (j < 3) {
    float acc = ab[j];
    for (int k = 0; k < DREAL; ++k) acc += gr[k] * aw[k * 3 + j];
    out[n * 3 + j] = acc;
  } else if (j < 9) {
    int jj = j - 3; float acc = sb[jj];
    for (int k = 0; k < DREAL; ++k) acc += gr[k] * sw[k * 6 + jj];
    out[3072 + n * 6 + jj] = dsigmoid(acc);
  } else if (j < 26) {
    int jj = j - 9; float acc = cb[jj];
    for (int k = 0; k < DREAL; ++k) acc += gr[k] * cw[k * 17 + jj];
    out[9216 + n * 17 + jj] = dsigmoid(acc);
  } else {
    int jj = j - 26; float acc = b2b[jj];
    for (int k = 0; k < 1024; ++k) acc += bx[(size_t)n * 1024 + k] * b2w[k * 4 + jj];
    out[26624 + n * 4 + jj] = acc;
  }
}

// ---------------------------------------------------------------------------
extern "C" void kernel_launch(void* const* d_in, const int* in_sizes, int n_in,
                              void* d_out, int out_size, void* d_ws, size_t ws_size,
                              hipStream_t stream)
{
  (void)in_sizes; (void)n_in; (void)out_size; (void)ws_size;
#define FIN(i) ((const float*)d_in[(i)])
  const float* features   = FIN(0);
  const int*   pair_idx   = (const int*)d_in[1];
  const float* union_feat = FIN(2);
  const float* spatial    = FIN(3);
  const int*   pred_lbl   = (const int*)d_in[4];
  const int layer_base[4] = {62, 26, 38, 50};  // local, global0..2

  char* base = (char*)d_ws; size_t off = 0;
  auto alloc = [&](size_t nb) -> void* {
    void* p = base + off; off += (nb + 255) & ~(size_t)255; return p; };

  // persistent f16 weights (~212 MB)
  h16* Wt_subj  = (h16*)alloc((size_t)512 * KFEATP * 2);
  h16* Wt_obj   = (h16*)alloc((size_t)512 * KFEATP * 2);
  h16* Wt_union = (h16*)alloc((size_t)256 * 1024 * 2);
  h16* Wt_conv2 = (h16*)alloc((size_t)256 * 1152 * 2);
  h16* Wt_vr    = (h16*)alloc((size_t)512 * 12544 * 2);
  h16 *Wt_qkv[4], *Wt_out[4], *Wt_ff1[4], *Wt_ff2[4];
  for (int l = 0; l < 4; ++l) {
    Wt_qkv[l] = (h16*)alloc((size_t)QKVP * DP * 2);
    Wt_out[l] = (h16*)alloc((size_t)DP * DP * 2);
    Wt_ff1[l] = (h16*)alloc((size_t)FFN * DP * 2);
    Wt_ff2[l] = (h16*)alloc((size_t)DP * FFN * 2);
  }
  h16* Wt_box1 = (h16*)alloc((size_t)1024 * DP * 2);

  // activations
  h16*   featA_s = (h16*)alloc((size_t)NPAIR * KFEATP * 2);
  h16*   featA_o = (h16*)alloc((size_t)NPAIR * KFEATP * 2);
  float* subj32  = (float*)alloc((size_t)NPAIR * 512 * 4);
  float* obj32   = (float*)alloc((size_t)NPAIR * 512 * 4);
  float* u32     = (float*)alloc((size_t)UROWS * 256 * 4);
  float* c2_32   = (float*)alloc((size_t)UROWS * 256 * 4);
  float* pooled  = (float*)alloc((size_t)1024 * 128 * 49 * 4);
  h16*   vrin16  = (h16*)alloc((size_t)NPAIR * 12544 * 2);
  float* vr32    = (float*)alloc((size_t)NPAIR * 512 * 4);
  float* bufA32  = (float*)alloc((size_t)NPAIR * DP * 4);
  h16*   bufA16  = (h16*)alloc((size_t)NPAIR * DP * 2);
  float* bufB32  = (float*)alloc((size_t)NPAIR * DP * 4);
  h16*   bufB16  = (h16*)alloc((size_t)NPAIR * DP * 2);
  float* box1_32 = (float*)alloc((size_t)NPAIR * 1024 * 4);

  // big arena, sequentially reused: uA16 -> conv1 out -> im2col -> encoder temps
  char* arena = (char*)alloc((size_t)116 * 1024 * 1024 + (1 << 20));
  h16*   uA16 = (h16*)arena;
  float* c1   = (float*)arena;
  h16*   im2  = (h16*)arena;
  size_t ao = 0;
  auto aall = [&](size_t nb) -> void* {
    void* p = arena + ao; ao += (nb + 255) & ~(size_t)255; return p; };
  float* qkv32   = (float*)aall((size_t)NPAIR * QKVP * 4);
  float* attnW   = (float*)aall((size_t)8 * NHEAD * 128 * 128 * 4);
  float* attnO32 = (float*)aall((size_t)NPAIR * DP * 4);
  h16*   attnO16 = (h16*)aall((size_t)NPAIR * DP * 2);
  float* proj32  = (float*)aall((size_t)NPAIR * DP * 4);
  float* h32buf  = (float*)aall((size_t)NPAIR * DP * 4);
  h16*   h16buf  = (h16*)aall((size_t)NPAIR * DP * 2);
  h16*   ff1_16  = (h16*)aall((size_t)NPAIR * FFN * 2);
  float* ff2_32  = (float*)aall((size_t)NPAIR * DP * 4);

  auto blocks = [](size_t n) { return dim3((unsigned)((n + 255) / 256)); };

  // ---- phase 0: weight conversion (transpose to (N,Kp) f16) ----
  k_cvt_wT<<<blocks((size_t)512 * KFEATP), 256, 0, stream>>>(FIN(81), Wt_subj, KFEAT, 512, KFEATP, 512);
  k_cvt_wT<<<blocks((size_t)512 * KFEATP), 256, 0, stream>>>(FIN(77), Wt_obj,  KFEAT, 512, KFEATP, 512);
  k_cvt_wT<<<blocks((size_t)256 * 1024),   256, 0, stream>>>(FIN(83), Wt_union, 1024, 256, 1024, 256);
  k_cvt_copy<<<blocks((size_t)256 * 1152), 256, 0, stream>>>(FIN(25), Wt_conv2, (size_t)256 * 1152);
  k_cvt_wT<<<blocks((size_t)512 * 12544),  256, 0, stream>>>(FIN(85), Wt_vr, 12544, 512, 12544, 512);
  for (int l = 0; l < 4; ++l) {
    int pb = layer_base[l];
    k_cvt_wT<<<blocks((size_t)QKVP * DP), 256, 0, stream>>>(FIN(pb + 11), Wt_qkv[l], DREAL, QKVN, DP, QKVP);
    k_cvt_wT<<<blocks((size_t)DP * DP),   256, 0, stream>>>(FIN(pb + 9),  Wt_out[l], DREAL, DREAL, DP, DP);
    k_cvt_wT<<<blocks((size_t)FFN * DP),  256, 0, stream>>>(FIN(pb + 1),  Wt_ff1[l], DREAL, FFN, DP, FFN);
    k_cvt_wT<<<blocks((size_t)DP * FFN),  256, 0, stream>>>(FIN(pb + 3),  Wt_ff2[l], FFN, DREAL, FFN, DP);
  }
  k_cvt_wT<<<blocks((size_t)1024 * DP), 256, 0, stream>>>(FIN(17), Wt_box1, DREAL, 1024, DP, 1024);

  // ---- phase 1: subj/obj projections ----
  k_gather<<<blocks((size_t)NPAIR * KFEATP), 256, 0, stream>>>(features, pair_idx, 0, featA_s);
  k_gather<<<blocks((size_t)NPAIR * KFEATP), 256, 0, stream>>>(features, pair_idx, 1, featA_o);
  k_gemm<<<dim3(8, 8), 128, 0, stream>>>(featA_s, Wt_subj, subj32, nullptr, FIN(80), KFEATP, 512, 512, 0);
  k_gemm<<<dim3(8, 8), 128, 0, stream>>>(featA_o, Wt_obj,  obj32,  nullptr, FIN(76), KFEATP, 512, 512, 0);

  // ---- phase 2: union einsum (uses arena as uA16) ----
  k_perm_union<<<blocks((size_t)UROWS * 1024), 256, 0, stream>>>(union_feat, uA16);
  k_gemm<<<dim3(4, UROWS / 128), 128, 0, stream>>>(uA16, Wt_union, u32, nullptr, FIN(82), 1024, 256, 256, 0);

  // ---- phase 3: conv path (arena: conv1 out, then im2col) ----
  k_conv1<<<blocks((size_t)1024 * 128 * 196), 256, 0, stream>>>(
      spatial, FIN(23), FIN(22), FIN(9), FIN(8), FIN(10), FIN(11), c1);
  k_pool<<<blocks((size_t)1024 * 128 * 49), 256, 0, stream>>>(c1, pooled);
  k_im2col<<<blocks((size_t)UROWS * 1152), 256, 0, stream>>>(pooled, im2);
  k_gemm<<<dim3(4, UROWS / 128), 128, 0, stream>>>(im2, Wt_conv2, c2_32, nullptr, FIN(24), 1152, 256, 256, 0);

  // ---- phase 4: vr ----
  k_vrin<<<blocks((size_t)1024 * 256 * 49), 256, 0, stream>>>(
      u32, c2_32, FIN(13), FIN(12), FIN(14), FIN(15), vrin16);
  k_gemm<<<dim3(8, 8), 128, 0, stream>>>(vrin16, Wt_vr, vr32, nullptr, FIN(84), 12544, 512, 512, 0);

  // ---- phase 5: concat into rel (bufA) ----
  k_concat<<<blocks((size_t)NPAIR * DP), 256, 0, stream>>>(
      subj32, obj32, vr32, FIN(74), FIN(75), pred_lbl, pair_idx, bufA32, bufA16);

  // ---- encoder layer ----
  auto enc = [&](const float* x32, const h16* x16, float* o32, h16* o16,
                 int li, int B, int S, int causal) {
    int pb = layer_base[li];
    k_gemm<<<dim3(QKVP / 64, 8), 128, 0, stream>>>(x16, Wt_qkv[li], qkv32, nullptr,
                                                   FIN(pb + 10), DP, QKVP, QKVN, 0);
    k_attn_sm<<<dim3((unsigned)(B * NHEAD * S)), dim3((unsigned)S), 0, stream>>>(qkv32, attnW, S, causal);
    k_attn_av<<<blocks((size_t)NPAIR * DP), 256, 0, stream>>>(qkv32, attnW, S, attnO32, attnO16);
    k_gemm<<<dim3(DP / 64, 8), 128, 0, stream>>>(attnO16, Wt_out[li], proj32, nullptr,
                                                 FIN(pb + 8), DP, DP, DREAL, 0);
    k_add_ln<<<dim3(NPAIR), 256, 0, stream>>>(x32, proj32, FIN(pb + 5), FIN(pb + 4), h32buf, h16buf);
    k_gemm<<<dim3(FFN / 64, 8), 128, 0, stream>>>(h16buf, Wt_ff1[li], nullptr, ff1_16,
                                                  FIN(pb + 0), DP, FFN, FFN, 1);
    k_gemm<<<dim3(DP / 64, 8), 128, 0, stream>>>(ff1_16, Wt_ff2[li], ff2_32, nullptr,
                                                 FIN(pb + 2), FFN, DP, DREAL, 0);
    k_add_ln<<<dim3(NPAIR), 256, 0, stream>>>(h32buf, ff2_32, FIN(pb + 7), FIN(pb + 6), o32, o16);
  };

  // local layer (B=128 frames, S=8), then global (B=8, S=128, causal)
  enc(bufA32, bufA16, bufB32, bufB16, 0, NFRAME, PPF, 0);
  k_tr_pe<<<blocks((size_t)NPAIR * DP), 256, 0, stream>>>(bufB32, bufA32, bufA16);
  enc(bufA32, bufA16, bufB32, bufB16, 1, PPF, NFRAME, 1);
  enc(bufB32, bufB16, bufA32, bufA16, 2, PPF, NFRAME, 1);
  enc(bufA32, bufA16, bufB32, bufB16, 3, PPF, NFRAME, 1);
  k_tr_back<<<blocks((size_t)NPAIR * DP), 256, 0, stream>>>(bufB32, bufA32, bufA16);  // g in bufA

  // ---- heads ----
  k_gemm<<<dim3(16, 8), 128, 0, stream>>>(bufA16, Wt_box1, box1_32, nullptr, FIN(16), DP, 1024, 1024, 1);
  k_heads<<<blocks((size_t)NPAIR * 30), 256, 0, stream>>>(
      bufA32, box1_32, FIN(7), FIN(6), FIN(79), FIN(78), FIN(21), FIN(20),
      FIN(19), FIN(18), (float*)d_out);
#undef FIN
}
